// Network_88699664597476
// MI455X (gfx1250) — matrix operand order repaired and checked by reading, not compiled, not run on hardware
//
#include <hip/hip_runtime.h>
#include <math.h>

#define BATCH 8
#define NPTS  4096

typedef __attribute__((ext_vector_type(16))) _Float16 v16h;
typedef __attribute__((ext_vector_type(8)))  float    v8f;

static inline int cdiv(int a, int b) { return (a + b - 1) / b; }

// ---------------------------------------------------------------------------
// prep: xyz0 = pcs with point 0 replaced by cp1
// ---------------------------------------------------------------------------
__global__ void prep_kernel(const float* __restrict__ pcs, const float* __restrict__ cp1,
                            float* __restrict__ xyz0, int total) {
  int g = blockIdx.x * blockDim.x + threadIdx.x;
  if (g >= total) return;
  int b   = g / (NPTS * 3);
  int rem = g % (NPTS * 3);
  int pt  = rem / 3;
  int k   = rem % 3;
  xyz0[g] = (pt == 0) ? cp1[b * 3 + k] : pcs[g];
}

// ---------------------------------------------------------------------------
// pack weights W[Kact][Nact] (f32 row-major) -> f16 WMMA B-fragment order:
// tile (kt,nt) is 512 halves, lane-major; lane l: n = nt*16+(l&15),
// element e: k = kt*32 + e + 16*(l>=16). Zero padded.
// featSwap>0 applies the K-permutation [feat(0..fs-1), xyz(3)] used by the
// SA input staging (feats first so async B128 LDS writes stay 16B aligned):
//   packed k < fs      -> source row 3+k   (feature weights)
//   packed k in [fs,fs+3) -> source row k-fs (xyz weights)
// ---------------------------------------------------------------------------
__global__ void pack_w_kernel(const float* __restrict__ W, _Float16* __restrict__ out,
                              int Kact, int Nact, int Kpad, int Npad, int featSwap,
                              int total) {
  int i = blockIdx.x * blockDim.x + threadIdx.x;
  if (i >= total) return;
  int e = i & 15;
  int l = (i >> 4) & 31;
  int t = i >> 9;
  int ntiles = Npad >> 4;
  int nt = t % ntiles, kt = t / ntiles;
  int n = nt * 16 + (l & 15);
  int k = kt * 32 + e + ((l >> 4) << 4);
  int ksrc;
  if (featSwap > 0) {
    if (k < featSwap)            ksrc = 3 + k;
    else if (k < featSwap + 3)   ksrc = k - featSwap;
    else                         ksrc = -1;
  } else {
    ksrc = (k < Kact) ? k : -1;
  }
  float v = (ksrc >= 0 && ksrc < Kact && n < Nact) ? W[ksrc * Nact + n] : 0.f;
  out[i] = (_Float16)v;
}

// ---------------------------------------------------------------------------
// Farthest point sampling: one workgroup (256 thr) per batch
// ---------------------------------------------------------------------------
__global__ void __launch_bounds__(256) fps_kernel(const float* __restrict__ xyz, int n,
                                                  int npoint, int* __restrict__ out) {
  int b = blockIdx.x, tid = threadIdx.x;
  const float* X = xyz + (size_t)b * n * 3;
  float md[16];
  {
    int c = 0;
    for (int t = tid; t < n; t += 256, ++c) md[c] = 1e10f;
  }
  __shared__ float sv[256];
  __shared__ int   si[256];
  __shared__ int   ssel;
  int sel = 0;
  for (int s = 0; s < npoint; ++s) {
    if (tid == 0) out[b * npoint + s] = sel;
    float lx = X[sel * 3], ly = X[sel * 3 + 1], lz = X[sel * 3 + 2];
    float best = -1e30f;
    int   bj   = 0x7fffffff;
    int c = 0;
    for (int t = tid; t < n; t += 256, ++c) {
      float dx = X[t * 3] - lx, dy = X[t * 3 + 1] - ly, dz = X[t * 3 + 2] - lz;
      float d = dx * dx + dy * dy + dz * dz;
      float m = md[c] < d ? md[c] : d;
      md[c] = m;
      if (m > best) { best = m; bj = t; }
    }
    sv[tid] = best; si[tid] = bj;
    __syncthreads();
    for (int o = 128; o > 0; o >>= 1) {
      if (tid < o) {
        float v2 = sv[tid + o]; int j2 = si[tid + o];
        if (v2 > sv[tid] || (v2 == sv[tid] && j2 < si[tid])) { sv[tid] = v2; si[tid] = j2; }
      }
      __syncthreads();
    }
    if (tid == 0) ssel = si[0];
    __syncthreads();
    sel = ssel;
    __syncthreads();
  }
}

// ---------------------------------------------------------------------------
// gather new_xyz = xyz[fidx]
// ---------------------------------------------------------------------------
__global__ void gather_xyz_kernel(const float* __restrict__ xyz, const int* __restrict__ fidx,
                                  float* __restrict__ nxyz, int n, int np, int total) {
  int g = blockIdx.x * blockDim.x + threadIdx.x;
  if (g >= total) return;
  int b = g / np;
  int j = fidx[g];
  const float* p = xyz + ((size_t)b * n + j) * 3;
  float* o = nxyz + (size_t)g * 3;
  o[0] = p[0]; o[1] = p[1]; o[2] = p[2];
}

// ---------------------------------------------------------------------------
// Ball query: first 32 in-radius indices in index order; pad with first hit
// (or 0 if none) -- matches the reference's sort trick.
// ---------------------------------------------------------------------------
__global__ void ballq_kernel(const float* __restrict__ xyz, int n,
                             const float* __restrict__ nxyz, int np, float r2,
                             int* __restrict__ idx, int total) {
  int g = blockIdx.x * blockDim.x + threadIdx.x;
  if (g >= total) return;
  int b = g / np;
  const float* X = xyz + (size_t)b * n * 3;
  float cx = nxyz[g * 3], cy = nxyz[g * 3 + 1], cz = nxyz[g * 3 + 2];
  int* o = idx + (size_t)g * 32;
  int cnt = 0;
  for (int j = 0; j < n && cnt < 32; ++j) {
    float dx = X[j * 3] - cx, dy = X[j * 3 + 1] - cy, dz = X[j * 3 + 2] - cz;
    if (dx * dx + dy * dy + dz * dz < r2) o[cnt++] = j;
  }
  int fill = cnt > 0 ? o[0] : 0;
  for (; cnt < 32; ++cnt) o[cnt] = fill;
}

// ---------------------------------------------------------------------------
// Per-wave WMMA GEMM tile: O(32,N) = relu(A(32,Kpad) @ Wpacked * sc + bi)
// Two M-tile accumulators so each B fragment is loaded once per K-tile.
// ---------------------------------------------------------------------------
__device__ __forceinline__ void wave_gemm_tile(const _Float16* __restrict__ A, int lda,
                                               _Float16* __restrict__ O, int ldo,
                                               const _Float16* __restrict__ wp,
                                               const float* __restrict__ sc,
                                               const float* __restrict__ bi,
                                               int Kpad, int N, int lane, int rowsValid) {
  const int ktiles = Kpad >> 5;
  const int ntiles = N >> 4;
  const int lhi = lane >> 4;   // 0 or 1
  const int llo = lane & 15;
  const _Float16* Arow0 = A + (size_t)llo * lda;
  const _Float16* Arow1 = A + (size_t)(16 + llo) * lda;
  for (int nt = 0; nt < ntiles; ++nt) {
    v8f acc0 = {};
    v8f acc1 = {};
    for (int kt = 0; kt < ktiles; ++kt) {
      const int kb = kt * 32 + (lhi << 4);
      v16h a0, a1;
#pragma unroll
      for (int e = 0; e < 16; ++e) {
        int k = kb + e;
        a0[e] = Arow0[k];
        a1[e] = Arow1[k];
      }
      v16h bv = *(const v16h*)(wp + (size_t)(kt * ntiles + nt) * 512 + (size_t)lane * 16);
      acc0 = __builtin_amdgcn_wmma_f32_16x16x32_f16(false, a0, false, bv,
                                                    (short)0, acc0, false, false);
      acc1 = __builtin_amdgcn_wmma_f32_16x16x32_f16(false, a1, false, bv,
                                                    (short)0, acc1, false, false);
    }
    const float scn = sc[nt * 16 + llo];
    const float bin = bi[nt * 16 + llo];
#pragma unroll
    for (int r = 0; r < 8; ++r) {
      int row0 = (lhi << 3) + r;
      float y0 = acc0[r] * scn + bin;
      y0 = y0 > 0.f ? y0 : 0.f;
      if (row0 < rowsValid) O[(size_t)row0 * ldo + nt * 16 + llo] = (_Float16)y0;
      int row1 = row0 + 16;
      float y1 = acc1[r] * scn + bin;
      y1 = y1 > 0.f ? y1 : 0.f;
      if (row1 < rowsValid) O[(size_t)row1 * ldo + nt * 16 + llo] = (_Float16)y1;
    }
  }
}

// ---------------------------------------------------------------------------
// Fused SA: 4 waves per block, one wave per (b, query point). Gather 32
// neighbors into the wave's LDS slice (features via async global->LDS DMA),
// 3 WMMA MLP layers ping-ponging LDS, then channel-wise max over samples.
// K layout for levels with features: [feat(0..featC-1), dxyz(3), pad].
// ---------------------------------------------------------------------------
__global__ void __launch_bounds__(128)
sa_kernel(const float* __restrict__ xyz, int n, const _Float16* __restrict__ feats, int featC,
          const float* __restrict__ nxyz, const int* __restrict__ bidx, int np,
          const _Float16* w0, const float* sc0, const float* bi0,
          const _Float16* w1, const float* sc1, const float* bi1,
          const _Float16* w2, const float* sc2, const float* bi2,
          int Kact0, int K0, int N0, int N1, int N2,
          _Float16* __restrict__ outf, int dmax) {
  extern __shared__ _Float16 smem[];
  const int wave = threadIdx.x >> 5;
  const int lane = threadIdx.x & 31;
  const int g = blockIdx.x * 4 + wave;
  const int b = g / np;
  _Float16* bufA = smem + (size_t)wave * 2 * 32 * dmax;
  _Float16* bufB = bufA + (size_t)32 * dmax;

  float cx = nxyz[(size_t)g * 3], cy = nxyz[(size_t)g * 3 + 1], cz = nxyz[(size_t)g * 3 + 2];
  int j = bidx[(size_t)g * 32 + lane];
  const float* p = xyz + ((size_t)b * n + j) * 3;
  _Float16* row = bufA + (size_t)lane * dmax;

  if (feats) {
    const _Float16* f = feats + ((size_t)b * n + j) * featC;
#if defined(__gfx1250__)
    unsigned loff = (unsigned)((const char*)row - (const char*)smem);
    for (int c = 0; c < featC; c += 8) {
      const _Float16* src = f + c;
      unsigned dst = loff + (unsigned)c * 2u;
      asm volatile("global_load_async_to_lds_b128 %0, %1, off"
                   :: "v"(dst), "v"(src) : "memory");
    }
#else
    for (int c = 0; c < featC; ++c) row[c] = f[c];
#endif
    row[featC + 0] = (_Float16)(p[0] - cx);
    row[featC + 1] = (_Float16)(p[1] - cy);
    row[featC + 2] = (_Float16)(p[2] - cz);
  } else {
    row[0] = (_Float16)(p[0] - cx);
    row[1] = (_Float16)(p[1] - cy);
    row[2] = (_Float16)(p[2] - cz);
    row[3] = (_Float16)p[0];
    row[4] = (_Float16)p[1];
    row[5] = (_Float16)p[2];
  }
  for (int c = Kact0; c < K0; ++c) row[c] = (_Float16)0.f;
#if defined(__gfx1250__)
  if (feats) asm volatile("s_wait_asynccnt 0x0" ::: "memory");
#endif
  __syncthreads();
  __builtin_prefetch(w0, 0, 1);
  wave_gemm_tile(bufA, dmax, bufB, dmax, w0, sc0, bi0, K0, N0, lane, 32);
  __syncthreads();
  wave_gemm_tile(bufB, dmax, bufA, dmax, w1, sc1, bi1, N0, N1, lane, 32);
  __syncthreads();
  wave_gemm_tile(bufA, dmax, bufB, dmax, w2, sc2, bi2, N1, N2, lane, 32);
  __syncthreads();
  // max over the 32 samples
  for (int c = lane; c < N2; c += 32) {
    float m = -1e30f;
    for (int s = 0; s < 32; ++s) {
      float v = (float)bufB[(size_t)s * dmax + c];
      m = v > m ? v : m;
    }
    outf[(size_t)g * N2 + c] = (_Float16)m;
  }
}

// ---------------------------------------------------------------------------
// Generic WMMA GEMM over row-tiles of 32: 8 waves per block, 1 tile per wave
// ---------------------------------------------------------------------------
__global__ void __launch_bounds__(256)
gemm_kernel(const _Float16* __restrict__ A, int Kpad,
            const _Float16* __restrict__ wp, const float* __restrict__ sc,
            const float* __restrict__ bi,
            _Float16* __restrict__ O, int N, int R) {
  int wave = threadIdx.x >> 5;
  int lane = threadIdx.x & 31;
  int tile = blockIdx.x * 8 + wave;
  if (tile * 32 >= R) return;   // wave-uniform exit, EXEC stays full for WMMA
  __builtin_prefetch(wp, 0, 1);
  const _Float16* At = A + (size_t)tile * 32 * Kpad;
  _Float16* Ot = O + (size_t)tile * 32 * N;
  int rows = R - tile * 32;
  if (rows > 32) rows = 32;
  wave_gemm_tile(At, Kpad, Ot, N, wp, sc, bi, Kpad, N, lane, rows);
}

// ---------------------------------------------------------------------------
// FP: 3-NN inverse-distance interpolation + concat, writes f16 A rows
// (padded K zeros). uf==nullptr -> uf is the query xyz itself (3 ch).
// ---------------------------------------------------------------------------
__global__ void fp_interp_kernel(const float* __restrict__ ux, int ux_bs,
                                 const float* __restrict__ kx, int Mk,
                                 const _Float16* __restrict__ uf, int Cu,
                                 const _Float16* __restrict__ kf, int Ck,
                                 _Float16* __restrict__ A, int Mu, int Kpad, int total) {
  int g = blockIdx.x * blockDim.x + threadIdx.x;
  if (g >= total) return;
  int b = g / Mu, u = g % Mu;
  const float* P = ux + (size_t)b * ux_bs + (size_t)u * 3;
  float px = P[0], py = P[1], pz = P[2];
  const float* KX = kx + (size_t)b * Mk * 3;
  float d0 = 1e30f, d1 = 1e30f, d2 = 1e30f;
  int i0 = 0, i1 = 0, i2 = 0;
  for (int j = 0; j < Mk; ++j) {
    float dx = KX[j * 3] - px, dy = KX[j * 3 + 1] - py, dz = KX[j * 3 + 2] - pz;
    float d = dx * dx + dy * dy + dz * dz;
    if (d < d0)      { d2 = d1; i2 = i1; d1 = d0; i1 = i0; d0 = d; i0 = j; }
    else if (d < d1) { d2 = d1; i2 = i1; d1 = d; i1 = j; }
    else if (d < d2) { d2 = d; i2 = j; }
  }
  float w0 = 1.f / (sqrtf(fmaxf(d0, 0.f)) + 1e-8f);
  float w1 = 1.f / (sqrtf(fmaxf(d1, 0.f)) + 1e-8f);
  float w2 = 1.f / (sqrtf(fmaxf(d2, 0.f)) + 1e-8f);
  float ws = w0 + w1 + w2;
  w0 /= ws; w1 /= ws; w2 /= ws;
  const _Float16* k0p = kf + ((size_t)b * Mk + i0) * Ck;
  const _Float16* k1p = kf + ((size_t)b * Mk + i1) * Ck;
  const _Float16* k2p = kf + ((size_t)b * Mk + i2) * Ck;
  _Float16* row = A + (size_t)g * Kpad;
  for (int c = 0; c < Ck; ++c)
    row[c] = (_Float16)(w0 * (float)k0p[c] + w1 * (float)k1p[c] + w2 * (float)k2p[c]);
  if (uf) {
    const _Float16* ur = uf + (size_t)g * Cu;
    for (int c = 0; c < Cu; ++c) row[Ck + c] = ur[c];
  } else {
    row[Ck] = (_Float16)px; row[Ck + 1] = (_Float16)py; row[Ck + 2] = (_Float16)pz;
  }
  for (int c = Ck + Cu; c < Kpad; ++c) row[c] = (_Float16)0.f;
}

// ---------------------------------------------------------------------------
// Tiny classifier head
// ---------------------------------------------------------------------------
__global__ void __launch_bounds__(128)
head_kernel(const _Float16* __restrict__ net1, const float* __restrict__ task,
            const float* __restrict__ cp1, const float* __restrict__ dir1,
            const float* __restrict__ tw, const float* __restrict__ tb,
            const float* __restrict__ cw, const float* __restrict__ cb,
            const float* __restrict__ dw, const float* __restrict__ db,
            const float* __restrict__ w1, const float* __restrict__ b1,
            const float* __restrict__ w2, const float* __restrict__ b2,
            float* __restrict__ out) {
  __shared__ float h[224];
  __shared__ float u[128];
  int b = blockIdx.x, t = threadIdx.x;
  h[t] = (float)net1[b * 128 + t];
  if (t < 32) {
    float a = tb[t];
    for (int j = 0; j < 3; ++j) a += task[b * 3 + j] * tw[j * 32 + t];
    h[128 + t] = a;
    float c = cb[t];
    for (int j = 0; j < 3; ++j) c += cp1[b * 3 + j] * cw[j * 32 + t];
    h[160 + t] = c;
    float d = db[t];
    for (int j = 0; j < 6; ++j) d += dir1[b * 6 + j] * dw[j * 32 + t];
    h[192 + t] = d;
  }
  __syncthreads();
  float a = b1[t];
  for (int j = 0; j < 224; ++j) a += h[j] * w1[j * 128 + t];
  u[t] = a > 0.f ? a : 0.01f * a;
  __syncthreads();
  if (t == 0) {
    float s = b2[0];
    for (int j = 0; j < 128; ++j) s += u[j] * w2[j];
    out[b] = 1.f / (1.f + expf(-s));
  }
}

// ---------------------------------------------------------------------------
// Host side
// ---------------------------------------------------------------------------
extern "C" void kernel_launch(void* const* d_in, const int* in_sizes, int n_in,
                              void* d_out, int out_size, void* d_ws, size_t ws_size,
                              hipStream_t stream) {
  (void)in_sizes; (void)n_in; (void)out_size; (void)ws_size;
  char* ws = (char*)d_ws;
  size_t off = 0;
  auto bump = [&](size_t bytes) -> char* {
    char* p = ws + off;
    off = (off + bytes + 255) & ~((size_t)255);
    return p;
  };

  // ---- scratch buffers -----------------------------------------------------
  static const int NPL[4] = {1024, 256, 64, 16};   // points produced per SA level
  static const int NL[4]  = {4096, 1024, 256, 64}; // points consumed per SA level
  static const int CL[4]  = {64, 128, 256, 512};   // channels produced per SA level

  float* xyz0 = (float*)bump((size_t)BATCH * NPTS * 3 * sizeof(float));
  int*      fidx[4];
  float*    nxyz[4];
  int*      bidx[4];
  _Float16* lf[4];
  for (int l = 0; l < 4; ++l) {
    fidx[l] = (int*)bump((size_t)BATCH * NPL[l] * sizeof(int));
    nxyz[l] = (float*)bump((size_t)BATCH * NPL[l] * 3 * sizeof(float));
    bidx[l] = (int*)bump((size_t)BATCH * NPL[l] * 32 * sizeof(int));
    lf[l]   = (_Float16*)bump((size_t)BATCH * NPL[l] * CL[l] * sizeof(_Float16));
  }
  _Float16* nf3   = (_Float16*)bump((size_t)BATCH * 64 * 256 * 2);
  _Float16* nf2   = (_Float16*)bump((size_t)BATCH * 256 * 256 * 2);
  _Float16* nf1   = (_Float16*)bump((size_t)BATCH * 1024 * 128 * 2);
  _Float16* nf0   = (_Float16*)bump((size_t)32 * 128 * 2);
  _Float16* fcout = (_Float16*)bump((size_t)32 * 128 * 2);
  _Float16* Abuf  = (_Float16*)bump((size_t)8192 * 320 * 2);
  _Float16* Obuf  = (_Float16*)bump((size_t)8192 * 256 * 2);

  // ---- weight packing ------------------------------------------------------
  struct LDsc { int wi, Kact, Kpad, N, fs; };
  static const LDsc L[22] = {
      {4, 6, 32, 32, 0},      {7, 32, 32, 32, 0},     {10, 32, 32, 64, 0},      // SA0
      {13, 67, 96, 64, 64},   {16, 64, 64, 64, 0},    {19, 64, 64, 128, 0},     // SA1
      {22, 131, 160, 128, 128}, {25, 128, 128, 128, 0}, {28, 128, 128, 256, 0}, // SA2
      {31, 259, 288, 256, 256}, {34, 256, 256, 256, 0}, {37, 256, 256, 512, 0}, // SA3
      {40, 131, 160, 128, 0}, {43, 128, 128, 128, 0}, {46, 128, 128, 128, 0},   // FP0
      {49, 320, 320, 256, 0}, {52, 256, 256, 128, 0},                           // FP1
      {55, 384, 384, 256, 0}, {58, 256, 256, 256, 0},                           // FP2
      {61, 768, 768, 256, 0}, {64, 256, 256, 256, 0},                           // FP3
      {67, 128, 128, 128, 0},                                                   // FC
  };
  size_t totHalves = 0;
  for (int i = 0; i < 22; ++i) totHalves += (size_t)L[i].Kpad * L[i].N;
  _Float16* wbase = (_Float16*)bump(totHalves * sizeof(_Float16));
  _Float16* wp[22];
  const float* sc[22];
  const float* bi[22];
  {
    size_t o = 0;
    for (int i = 0; i < 22; ++i) {
      wp[i] = wbase + o;
      o += (size_t)L[i].Kpad * L[i].N;
      sc[i] = (const float*)d_in[L[i].wi + 1];
      bi[i] = (const float*)d_in[L[i].wi + 2];
      int tot = L[i].Kpad * L[i].N;
      pack_w_kernel<<<cdiv(tot, 256), 256, 0, stream>>>(
          (const float*)d_in[L[i].wi], wp[i], L[i].Kact, L[i].N, L[i].Kpad, L[i].N,
          L[i].fs, tot);
    }
  }

  // ---- prep ----------------------------------------------------------------
  {
    int tot = BATCH * NPTS * 3;
    prep_kernel<<<cdiv(tot, 256), 256, 0, stream>>>((const float*)d_in[0],
                                                    (const float*)d_in[2], xyz0, tot);
  }

  // ---- SA levels -----------------------------------------------------------
  static const float R2[4]    = {0.01f, 0.04f, 0.16f, 0.64f};
  static const int   KACT0[4] = {6, 67, 131, 259};
  static const int   K0[4]    = {32, 96, 160, 288};
  static const int   N0_[4]   = {32, 64, 128, 256};
  static const int   N1_[4]   = {32, 64, 128, 256};
  static const int   N2_[4]   = {64, 128, 256, 512};
  static const int   DMAX[4]  = {64, 128, 256, 512};
  static const int   LB[4]    = {0, 3, 6, 9};

  for (int l = 0; l < 4; ++l) {
    const float* xl = (l == 0) ? xyz0 : nxyz[l - 1];
    int n = NL[l], np = NPL[l];
    fps_kernel<<<BATCH, 256, 0, stream>>>(xl, n, np, fidx[l]);
    gather_xyz_kernel<<<cdiv(BATCH * np, 256), 256, 0, stream>>>(xl, fidx[l], nxyz[l], n, np,
                                                                 BATCH * np);
    ballq_kernel<<<cdiv(BATCH * np, 128), 128, 0, stream>>>(xl, n, nxyz[l], np, R2[l], bidx[l],
                                                            BATCH * np);
    const _Float16* fl = (l == 0) ? nullptr : lf[l - 1];
    int lb = LB[l];
    sa_kernel<<<(BATCH * np) / 4, 128, (size_t)512 * DMAX[l], stream>>>(
        xl, n, fl, (l == 0) ? 0 : CL[l - 1], nxyz[l], bidx[l], np,
        wp[lb], sc[lb], bi[lb], wp[lb + 1], sc[lb + 1], bi[lb + 1],
        wp[lb + 2], sc[lb + 2], bi[lb + 2],
        KACT0[l], K0[l], N0_[l], N1_[l], N2_[l], lf[l], DMAX[l]);
  }

  auto gemm = [&](const _Float16* A, int Kpad, int i, _Float16* O, int N, int R) {
    int tiles = cdiv(R, 32);
    gemm_kernel<<<cdiv(tiles, 8), 256, 0, stream>>>(A, Kpad, wp[i], sc[i], bi[i], O, N, R);
  };

  // ---- FP stages (i = 4..2 full, i = 1 only at point 0) --------------------
  // i=4: (64 pts) interp from level4, concat lf[2] -> 768 -> 256 -> 256
  fp_interp_kernel<<<cdiv(512, 128), 128, 0, stream>>>(nxyz[2], NPL[2] * 3, nxyz[3], 16,
                                                       lf[2], 256, lf[3], 512, Abuf, 64, 768, 512);
  gemm(Abuf, 768, 19, Obuf, 256, 512);
  gemm(Obuf, 256, 20, nf3, 256, 512);

  // i=3: (256 pts) interp from nf3, concat lf[1] -> 384 -> 256 -> 256
  fp_interp_kernel<<<cdiv(2048, 128), 128, 0, stream>>>(nxyz[1], NPL[1] * 3, nxyz[2], 64,
                                                        lf[1], 128, nf3, 256, Abuf, 256, 384, 2048);
  gemm(Abuf, 384, 17, Obuf, 256, 2048);
  gemm(Obuf, 256, 18, nf2, 256, 2048);

  // i=2: (1024 pts) interp from nf2, concat lf[0] -> 320 -> 256 -> 128
  fp_interp_kernel<<<cdiv(8192, 128), 128, 0, stream>>>(nxyz[0], NPL[0] * 3, nxyz[1], 256,
                                                        lf[0], 64, nf2, 256, Abuf, 1024, 320, 8192);
  gemm(Abuf, 320, 15, Obuf, 256, 8192);
  gemm(Obuf, 256, 16, nf1, 128, 8192);

  // i=1: only point 0 matters downstream (feat0[:,0,:]) -> 8 rows total
  fp_interp_kernel<<<1, 128, 0, stream>>>(xyz0, NPTS * 3, nxyz[0], 1024,
                                          nullptr, 3, nf1, 128, Abuf, 1, 160, BATCH);
  gemm(Abuf, 160, 12, Obuf, 128, BATCH);
  gemm(Obuf, 128, 13, Abuf, 128, BATCH);
  gemm(Abuf, 128, 14, nf0, 128, BATCH);

  // fc: feat0 = relu(nf0 @ Wfc * sc + bi), 8 rows
  gemm(nf0, 128, 21, fcout, 128, BATCH);

  // head
  head_kernel<<<BATCH, 128, 0, stream>>>(
      fcout, (const float*)d_in[1], (const float*)d_in[2], (const float*)d_in[3],
      (const float*)d_in[70], (const float*)d_in[71], (const float*)d_in[72],
      (const float*)d_in[73], (const float*)d_in[74], (const float*)d_in[75],
      (const float*)d_in[76], (const float*)d_in[77], (const float*)d_in[78],
      (const float*)d_in[79], (float*)d_out);
}
